// Dnn_with_Attention_87591563034710
// MI455X (gfx1250) — compile-verified, matrix operand order checked by reading, not run on hardware
//
#include <hip/hip_runtime.h>
#include <hip/hip_bf16.h>

// ---------------------------------------------------------------------------
// Problem constants (from the reference)
// ---------------------------------------------------------------------------
#define TOTAL_N 131072
#define FEAT    78
#define K1P     128       // FEAT padded up to multiple of 64 (TK) for WMMA
#define HID     1024
#define SEGS    256
#define NCLS    10

typedef __attribute__((ext_vector_type(16))) _Float16 v16h;
typedef __attribute__((ext_vector_type(8)))  _Float16 v8h;
typedef __attribute__((ext_vector_type(8)))  float    v8f;

union HalfVec16 { v16h v; v8h h[2]; };

// ---------------------------------------------------------------------------
// CDNA5 async global->LDS copy (ASYNCcnt-tracked, bypasses VGPR staging).
// LDS operand is the low 32 bits of the generic pointer (aperture rule:
// the LDS address is the pointer's low dword).
// ---------------------------------------------------------------------------
__device__ __forceinline__ void async_copy_b128(void* lds_ptr, const void* gptr)
{
    unsigned lds_off = (unsigned)(unsigned long long)lds_ptr;
    asm volatile("global_load_async_to_lds_b128 %0, %1, off"
                 :: "v"(lds_off), "v"(gptr)
                 : "memory");
}

__device__ __forceinline__ void wait_async0()
{
    asm volatile("s_wait_asynccnt 0x0" ::: "memory");
}

// ---------------------------------------------------------------------------
// Convert + pad x (fp32 [N, FEAT]) -> Xp (f16 [N, K1P]), zero pad cols >= FEAT
// ---------------------------------------------------------------------------
__global__ __launch_bounds__(256) void cvt_x_pad_kernel(
    const float* __restrict__ x, _Float16* __restrict__ Xp, int M)
{
    long idx = (long)blockIdx.x * blockDim.x + threadIdx.x;
    long total = (long)M * K1P;
    if (idx >= total) return;
    int k = (int)(idx % K1P);
    long r = idx / K1P;
    Xp[idx] = (k < FEAT) ? (_Float16)x[r * FEAT + k] : (_Float16)0.0f;
}

// ---------------------------------------------------------------------------
// Convert + transpose W (fp32 [K, N]) -> Wt (f16 [N, Kp]), zero pad k >= K
// ---------------------------------------------------------------------------
__global__ __launch_bounds__(256) void cvt_w_t_kernel(
    const float* __restrict__ W, _Float16* __restrict__ Wt, int K, int N, int Kp)
{
    long idx = (long)blockIdx.x * blockDim.x + threadIdx.x;
    long total = (long)N * Kp;
    if (idx >= total) return;
    int k = (int)(idx % Kp);
    long n = idx / Kp;
    Wt[idx] = (k < K) ? (_Float16)W[(long)k * N + n] : (_Float16)0.0f;
}

// ---------------------------------------------------------------------------
// Fused GEMM + bias + ReLU:   C[M,N] (f16) = relu(A[M,K] @ Bt[N,K]^T + bias)
//   A  : f16 row-major, stride K (K multiple of 64, M multiple of 128)
//   Bt : f16 row-major [N,K] (transposed weights), N multiple of 128
// Block tile 128x128, 8 wave32 waves in 2(M) x 4(N) grid; each wave owns a
// 64x32 region = 4x2 WMMA 16x16x32 tiles with fp32 accumulators.
// K-stage = 64 (2 WMMA K-substeps -> 16 WMMAs per barrier), double-buffered
// LDS filled with global_load_async_to_lds_b128.
// ---------------------------------------------------------------------------
#define TM 128
#define TN 128
#define TK 64

__global__ __launch_bounds__(256) void gemm_bias_relu_f16_kernel(
    const _Float16* __restrict__ A,
    const _Float16* __restrict__ Bt,
    const float*    __restrict__ bias,
    _Float16*       __restrict__ C,
    int M, int N, int K)
{
    __shared__ __align__(16) _Float16 As[2][TM][TK];   // 2 x 16 KB
    __shared__ __align__(16) _Float16 Bs[2][TN][TK];   // 2 x 16 KB

    const int tid  = threadIdx.x;
    const int lane = tid & 31;
    const int wave = tid >> 5;      // 0..7
    const int wm   = wave >> 2;     // 0..1 -> M offset 64*wm inside tile
    const int wn   = wave & 3;      // 0..3 -> N offset 32*wn inside tile

    const long blockM = (long)blockIdx.x * TM;
    const long blockN = (long)blockIdx.y * TN;

    v8f acc[4][2];
    const v8f vzero = {0.f, 0.f, 0.f, 0.f, 0.f, 0.f, 0.f, 0.f};
#pragma unroll
    for (int i = 0; i < 4; ++i)
#pragma unroll
        for (int j = 0; j < 2; ++j) acc[i][j] = vzero;

    const int fr = lane & 15;          // fragment row (M or N inside 16-tile)
    const int kb = (lane >> 4) * 8;    // K sub-block select per ISA layout
    const int rowAdd = (lane >> 4) * 8;

    // Async-stage one 128x64 A tile + 128x64 B tile into LDS buffer `buf`.
    // 128 rows x 8 b128-chunks = 1024 chunks per tile; 256 threads x 4.
    auto stage = [&](int buf, int k0) {
#pragma unroll
        for (int p = 0; p < 4; ++p) {
            const int c   = tid + p * 256;
            const int row = c >> 3;          // 0..127
            const int col = (c & 7) * 8;     // halves
            async_copy_b128(&As[buf][row][col],
                            &A [(blockM + row) * (long)K + k0 + col]);
            async_copy_b128(&Bs[buf][row][col],
                            &Bt[(blockN + row) * (long)K + k0 + col]);
        }
    };

    const int nstages = K / TK;

    stage(0, 0);
    wait_async0();
    __syncthreads();

    for (int s = 0; s < nstages; ++s) {
        const int buf = s & 1;
        if (s + 1 < nstages) stage(buf ^ 1, (s + 1) * TK);

        // ---- compute: 2 K-substeps of 32, 8 WMMAs each ----
#pragma unroll
        for (int ks = 0; ks < TK; ks += 32) {
            v16h afrag[4], bfrag[2];
#pragma unroll
            for (int i = 0; i < 4; ++i) {
                const int r = wm * 64 + i * 16 + fr;
                HalfVec16 u;
                u.h[0] = *(const v8h*)&As[buf][r][ks + kb];
                u.h[1] = *(const v8h*)&As[buf][r][ks + kb + 16];
                afrag[i] = u.v;
            }
#pragma unroll
            for (int j = 0; j < 2; ++j) {
                const int r = wn * 32 + j * 16 + fr;
                HalfVec16 u;
                u.h[0] = *(const v8h*)&Bs[buf][r][ks + kb];
                u.h[1] = *(const v8h*)&Bs[buf][r][ks + kb + 16];
                bfrag[j] = u.v;
            }
#pragma unroll
            for (int i = 0; i < 4; ++i)
#pragma unroll
                for (int j = 0; j < 2; ++j)
                    acc[i][j] = __builtin_amdgcn_wmma_f32_16x16x32_f16(
                        false, afrag[i], false, bfrag[j],
                        (short)0, acc[i][j], false, false);
        }

        // publish next buffer: own async writes done, then block-wide barrier
        wait_async0();
        __syncthreads();
    }

    // ---- epilogue: bias + ReLU, store f16 (C/D layout: VGPR r -> M=r / r+8) --
#pragma unroll
    for (int j = 0; j < 2; ++j) {
        const long col = blockN + wn * 32 + j * 16 + fr;
        const float bc = bias[col];
#pragma unroll
        for (int i = 0; i < 4; ++i) {
            const long rowBase = blockM + wm * 64 + i * 16 + rowAdd;
#pragma unroll
            for (int r = 0; r < 8; ++r) {
                float v = acc[i][j][r] + bc;
                v = fmaxf(v, 0.0f);
                C[(rowBase + r) * (long)N + col] = (_Float16)v;
            }
        }
    }
}

// ---------------------------------------------------------------------------
// Attention logits: scores[i] = relu(dot(H[i,:], W5) + b5), one wave per row
// ---------------------------------------------------------------------------
__global__ __launch_bounds__(256) void scores_kernel(
    const _Float16* __restrict__ H, const float* __restrict__ W5,
    const float* __restrict__ b5, float* __restrict__ scores, int M, int K)
{
    const int lane = threadIdx.x & 31;
    const int row  = blockIdx.x * (blockDim.x >> 5) + (threadIdx.x >> 5);
    if (row >= M) return;
    float s = 0.f;
    const _Float16* hp = H + (long)row * K;
    for (int k = lane; k < K; k += 32) s += (float)hp[k] * W5[k];
#pragma unroll
    for (int off = 16; off > 0; off >>= 1) s += __shfl_down(s, off, 32);
    if (lane == 0) scores[row] = fmaxf(s + b5[0], 0.0f);
}

// ---------------------------------------------------------------------------
// Exclusive prefix sum of lengths -> segment start offsets (tiny, 1 thread)
// ---------------------------------------------------------------------------
__global__ void seg_offsets_kernel(const int* __restrict__ lens,
                                   int* __restrict__ offs, int n)
{
    if (blockIdx.x == 0 && threadIdx.x == 0) {
        int acc = 0;
        for (int i = 0; i < n; ++i) { offs[i] = acc; acc += lens[i]; }
    }
}

// ---------------------------------------------------------------------------
// Per-segment softmax over scores + weighted-sum pooling of H -> f16 [SEG,HID]
// One block per segment; 256 threads; 4 columns per thread.
// ---------------------------------------------------------------------------
__global__ __launch_bounds__(256) void attn_pool_kernel(
    const _Float16* __restrict__ H, const float* __restrict__ scores,
    const int* __restrict__ offs, const int* __restrict__ lens,
    _Float16* __restrict__ pooledh, int K)
{
    const int seg = blockIdx.x;
    const int off = offs[seg];
    const int len = lens[seg];
    const int tid = threadIdx.x;
    __shared__ float red[256];

    // pass 1: segment max (stable softmax)
    float m = -1e30f;
    for (int i = tid; i < len; i += 256) m = fmaxf(m, scores[off + i]);
    red[tid] = m; __syncthreads();
    for (int s = 128; s > 0; s >>= 1) {
        if (tid < s) red[tid] = fmaxf(red[tid], red[tid + s]);
        __syncthreads();
    }
    m = red[0]; __syncthreads();

    // pass 2: exp-sum
    float sum = 0.f;
    for (int i = tid; i < len; i += 256) sum += __expf(scores[off + i] - m);
    red[tid] = sum; __syncthreads();
    for (int s = 128; s > 0; s >>= 1) {
        if (tid < s) red[tid] += red[tid + s];
        __syncthreads();
    }
    const float inv = 1.0f / red[0];

    // pass 3: weighted accumulate, 4 columns per thread
    const int c0 = tid * 4;
    float a0 = 0.f, a1 = 0.f, a2 = 0.f, a3 = 0.f;
    for (int i = 0; i < len; ++i) {
        const float w = __expf(scores[off + i] - m) * inv;
        const _Float16* hp = &H[(long)(off + i) * K + c0];
        a0 += w * (float)hp[0];
        a1 += w * (float)hp[1];
        a2 += w * (float)hp[2];
        a3 += w * (float)hp[3];
    }
    _Float16* op = &pooledh[(long)seg * K + c0];
    op[0] = (_Float16)a0; op[1] = (_Float16)a1;
    op[2] = (_Float16)a2; op[3] = (_Float16)a3;
}

// ---------------------------------------------------------------------------
// Final head: out[row, c] = dot(T[row,:], W7[:,c]) + b7[c]   (no ReLU)
// One block per row.
// ---------------------------------------------------------------------------
__global__ __launch_bounds__(256) void final_out_kernel(
    const _Float16* __restrict__ T, const float* __restrict__ W7,
    const float* __restrict__ b7, float* __restrict__ out, int K)
{
    const int row = blockIdx.x;
    const int tid = threadIdx.x;
    __shared__ float red[256];

    float acc[NCLS];
#pragma unroll
    for (int c = 0; c < NCLS; ++c) acc[c] = 0.f;

    for (int k = tid; k < K; k += 256) {
        const float h = (float)T[(long)row * K + k];
        const float* wr = &W7[(long)k * NCLS];
#pragma unroll
        for (int c = 0; c < NCLS; ++c) acc[c] += h * wr[c];
    }
#pragma unroll
    for (int c = 0; c < NCLS; ++c) {
        red[tid] = acc[c]; __syncthreads();
        for (int s = 128; s > 0; s >>= 1) {
            if (tid < s) red[tid] += red[tid + s];
            __syncthreads();
        }
        if (tid == 0) out[row * NCLS + c] = red[0] + b7[c];
        __syncthreads();
    }
}

// ---------------------------------------------------------------------------
// Host launcher
// Inputs (setup_inputs order):
//  0:x 1:W1 2:b1 3:W2 4:b2 5:W3 6:b3 7:W4 8:b4 9:W5 10:b5 11:W6 12:b6
//  13:W7 14:b7 15:lengths
// ---------------------------------------------------------------------------
extern "C" void kernel_launch(void* const* d_in, const int* in_sizes, int n_in,
                              void* d_out, int out_size, void* d_ws, size_t ws_size,
                              hipStream_t stream)
{
    const float* x   = (const float*)d_in[0];
    const float* W1  = (const float*)d_in[1];
    const float* b1  = (const float*)d_in[2];
    const float* W2  = (const float*)d_in[3];
    const float* b2  = (const float*)d_in[4];
    const float* W3  = (const float*)d_in[5];
    const float* b3  = (const float*)d_in[6];
    const float* W4  = (const float*)d_in[7];
    const float* b4  = (const float*)d_in[8];
    const float* W5  = (const float*)d_in[9];
    const float* b5  = (const float*)d_in[10];
    const float* W6  = (const float*)d_in[11];
    const float* b6  = (const float*)d_in[12];
    const float* W7  = (const float*)d_in[13];
    const float* b7  = (const float*)d_in[14];
    const int*   lengths = (const int*)d_in[15];
    float* out = (float*)d_out;

    // --- workspace carve-out (256-byte aligned blocks) ---
    char* base = (char*)d_ws;
    size_t off = 0;
    auto take = [&](size_t bytes) -> void* {
        void* p = base + off;
        off = (off + bytes + 255) & ~(size_t)255;
        return p;
    };
    _Float16* Xp      = (_Float16*)take((size_t)TOTAL_N * K1P * 2);
    _Float16* W1t     = (_Float16*)take((size_t)HID * K1P * 2);
    _Float16* W2t     = (_Float16*)take((size_t)HID * HID * 2);
    _Float16* W3t     = (_Float16*)take((size_t)HID * HID * 2);
    _Float16* W4t     = (_Float16*)take((size_t)HID * HID * 2);
    _Float16* W6t     = (_Float16*)take((size_t)HID * HID * 2);
    _Float16* Ha      = (_Float16*)take((size_t)TOTAL_N * HID * 2);
    _Float16* Hb      = (_Float16*)take((size_t)TOTAL_N * HID * 2);
    float*    scores  = (float*)   take((size_t)TOTAL_N * 4);
    int*      offs    = (int*)     take((size_t)SEGS * 4);
    _Float16* pooledh = (_Float16*)take((size_t)SEGS * HID * 2);
    _Float16* Th      = (_Float16*)take((size_t)SEGS * HID * 2);

    // --- precision conversion / layout prep ---
    {
        long total = (long)TOTAL_N * K1P;
        cvt_x_pad_kernel<<<(unsigned)((total + 255) / 256), 256, 0, stream>>>(x, Xp, TOTAL_N);
    }
    {
        long t1 = (long)HID * K1P;
        cvt_w_t_kernel<<<(unsigned)((t1 + 255) / 256), 256, 0, stream>>>(W1, W1t, FEAT, HID, K1P);
        long t2 = (long)HID * HID;
        unsigned g2 = (unsigned)((t2 + 255) / 256);
        cvt_w_t_kernel<<<g2, 256, 0, stream>>>(W2, W2t, HID, HID, HID);
        cvt_w_t_kernel<<<g2, 256, 0, stream>>>(W3, W3t, HID, HID, HID);
        cvt_w_t_kernel<<<g2, 256, 0, stream>>>(W4, W4t, HID, HID, HID);
        cvt_w_t_kernel<<<g2, 256, 0, stream>>>(W6, W6t, HID, HID, HID);
    }

    // --- frame MLP: 4 fused WMMA GEMM layers ---
    dim3 gBig(TOTAL_N / TM, HID / TN);
    gemm_bias_relu_f16_kernel<<<gBig, 256, 0, stream>>>(Xp, W1t, b1, Ha, TOTAL_N, HID, K1P);
    gemm_bias_relu_f16_kernel<<<gBig, 256, 0, stream>>>(Ha, W2t, b2, Hb, TOTAL_N, HID, HID);
    gemm_bias_relu_f16_kernel<<<gBig, 256, 0, stream>>>(Hb, W3t, b3, Ha, TOTAL_N, HID, HID);
    gemm_bias_relu_f16_kernel<<<gBig, 256, 0, stream>>>(Ha, W4t, b4, Hb, TOTAL_N, HID, HID);
    // H4 now lives in Hb (f16)

    // --- attention logits + segment softmax pooling ---
    scores_kernel<<<TOTAL_N / 8, 256, 0, stream>>>(Hb, W5, b5, scores, TOTAL_N, HID);
    seg_offsets_kernel<<<1, 32, 0, stream>>>(lengths, offs, SEGS);
    attn_pool_kernel<<<SEGS, 256, 0, stream>>>(Hb, scores, offs, lengths, pooledh, HID);

    // --- segment head: relu(pooled @ W6 + b6) via WMMA, then final linear ---
    dim3 gSeg(SEGS / TM, HID / TN);
    gemm_bias_relu_f16_kernel<<<gSeg, 256, 0, stream>>>(pooledh, W6t, b6, Th, SEGS, HID, HID);
    final_out_kernel<<<SEGS, 256, 0, stream>>>(Th, W7, b7, out, HID);
}
